// LLM_61521111548413
// MI455X (gfx1250) — compile-verified, matrix operand order checked by reading, not run on hardware
//
#include <hip/hip_runtime.h>
#include <hip/hip_bf16.h>
#include <stdint.h>

// ---- problem constants (from reference setup) ----
#define DDIM   768           // hidden dim (K)
#define MROWS  4096          // B*S = 2*2048 (M)
#define VROWS  50257         // vocab (N)
#define VPAD   50304         // 393 * 128, padded N for tiling
#define EPSF   1e-5f

typedef __attribute__((ext_vector_type(16))) __bf16 v16bf;
typedef __attribute__((ext_vector_type(8)))  __bf16 v8bf;
typedef __attribute__((ext_vector_type(4)))  __bf16 v4bf;
typedef __attribute__((ext_vector_type(8)))  float  v8f;

__device__ __forceinline__ __bf16 f32_to_bf16_rne(float f) {
    union { float f; uint32_t u; } a; a.f = f;
    uint32_t r = (a.u + 0x7fffu + ((a.u >> 16) & 1u)) >> 16;
    union { uint16_t s; __bf16 b; } o; o.s = (uint16_t)r;
    return o.b;
}

// Kernel 1: h_bf16[row] = rmsnorm(embedding[token[row]]) * w, converted to bf16.
// One 256-thread block per row; 768 = 3 * 256 elements per row.
__global__ void __launch_bounds__(256) k_gather_rmsnorm_bf16(
    const int* __restrict__ tokens, const float* __restrict__ emb,
    const float* __restrict__ w, __bf16* __restrict__ hb)
{
    const int row = blockIdx.x;
    const int tid = threadIdx.x;
    const int tok = tokens[row];
    const float* x = emb + (size_t)tok * DDIM;

    float xv[3];
    float ss = 0.f;
    #pragma unroll
    for (int i = 0; i < 3; ++i) { xv[i] = x[tid + 256 * i]; ss += xv[i] * xv[i]; }

    __shared__ float red[256];
    red[tid] = ss;
    __syncthreads();
    #pragma unroll
    for (int s = 128; s > 0; s >>= 1) {
        if (tid < s) red[tid] += red[tid + s];
        __syncthreads();
    }
    const float scale = rsqrtf(red[0] * (1.0f / (float)DDIM) + EPSF);

    __bf16* o = hb + (size_t)row * DDIM;
    #pragma unroll
    for (int i = 0; i < 3; ++i) {
        const int c = tid + 256 * i;
        o[c] = f32_to_bf16_rne(xv[i] * scale * w[c]);
    }
}

// Kernel 2: convert output_embedding fp32 -> bf16 into padded [VPAD][DDIM];
// pad rows written as zeros every call (fully deterministic overwrite of ws).
__global__ void __launch_bounds__(256) k_convert_w_bf16(
    const float* __restrict__ wsrc, __bf16* __restrict__ wdst)
{
    const size_t idx4 = ((size_t)blockIdx.x * 256 + threadIdx.x) * 4;
    const size_t total = (size_t)VPAD * DDIM;
    if (idx4 >= total) return;
    const size_t valid = (size_t)VROWS * DDIM;   // multiple of 4
    v4bf o;
    if (idx4 < valid) {
        const float4 f = *(const float4*)(wsrc + idx4);
        o.x = f32_to_bf16_rne(f.x);
        o.y = f32_to_bf16_rne(f.y);
        o.z = f32_to_bf16_rne(f.z);
        o.w = f32_to_bf16_rne(f.w);
    } else {
        o.x = o.y = o.z = o.w = (__bf16)0.0f;
    }
    *(v4bf*)(wdst + idx4) = o;
}

// Kernel 3: logits[M][V] = h_bf [M][K] x Wb[V][K]^T, bf16 WMMA, fp32 accum.
// Block tile 256(M) x 128(N); 8 waves in a 4x2 grid; each wave: 64x64 via a
// 4x4 grid of v_wmma_f32_16x16x32_bf16 accumulators (128 VGPRs).
// Fragments loaded straight from global (192MB L2 holds the 77MB bf16 W).
// All addressing through 32-bit byte offsets off uniform bases so the
// backend can use saddr+voffset (GVS) loads + 32-bit offset increments,
// keeping offset registers disjoint from fragment registers (avoids the
// 4-NOP WMMA->VALU WAR hazard).
__global__ void __launch_bounds__(256, 1) k_gemm_bf16_wmma(
    const __bf16* __restrict__ A,    // [MROWS][DDIM]
    const __bf16* __restrict__ Bm,   // [VPAD][DDIM]
    float* __restrict__ Cout)        // [MROWS][VROWS]
{
    const int lane = threadIdx.x & 31;
    const int wave = threadIdx.x >> 5;
    const int lr   = lane & 15;      // row-within-tile for A/B fragments
    const int hi   = lane >> 4;      // K-half: 0 -> K{0..7,16..23}, 1 -> K{8..15,24..31}
    const int wm   = wave >> 1;      // 0..3
    const int wn   = wave & 1;       // 0..1
    const int m0   = blockIdx.y * 256 + wm * 64;
    const int n0   = blockIdx.x * 128 + wn * 64;

    v8f acc[4][4] = {};

    // 32-bit byte offsets (A spans 6.3MB, Bm spans 77.3MB: both < 4GB)
    uint32_t aOff[4], bOff[4];
    #pragma unroll
    for (int mi = 0; mi < 4; ++mi)
        aOff[mi] = (uint32_t)((m0 + mi * 16 + lr) * DDIM + hi * 8) * 2u;
    #pragma unroll
    for (int ni = 0; ni < 4; ++ni)
        bOff[ni] = (uint32_t)((n0 + ni * 16 + lr) * DDIM + hi * 8) * 2u;

    const char* Ab = (const char*)A;
    const char* Bb = (const char*)Bm;

    union Frag { v16bf v; v8bf h[2]; };

    #pragma unroll 1
    for (int k = 0; k < DDIM; k += 32) {
        v16bf a[4], b[4];
        #pragma unroll
        for (int mi = 0; mi < 4; ++mi) {
            Frag f;
            f.h[0] = *(const v8bf*)(Ab + aOff[mi]);        // K{0..7} / {8..15}
            f.h[1] = *(const v8bf*)(Ab + aOff[mi] + 32);   // K{16..23} / {24..31}
            a[mi] = f.v;
        }
        #pragma unroll
        for (int ni = 0; ni < 4; ++ni) {
            Frag f;
            f.h[0] = *(const v8bf*)(Bb + bOff[ni]);
            f.h[1] = *(const v8bf*)(Bb + bOff[ni] + 32);
            b[ni] = f.v;
        }
        #pragma unroll
        for (int mi = 0; mi < 4; ++mi)
            #pragma unroll
            for (int ni = 0; ni < 4; ++ni)
                acc[mi][ni] = __builtin_amdgcn_wmma_f32_16x16x32_bf16(
                    false, a[mi], false, b[ni], (short)0, acc[mi][ni],
                    false, false);
        #pragma unroll
        for (int mi = 0; mi < 4; ++mi) aOff[mi] += 64u;    // 32 bf16 = 64B
        #pragma unroll
        for (int ni = 0; ni < 4; ++ni) bOff[ni] += 64u;
    }

    // Store: accumulator VGPR j holds M = j (lanes 0-15) / 8+j (lanes 16-31),
    // N = lane & 15.  Only the last N-column block touches the pad; n0 is
    // wave-uniform so this branch is scalar.
    char* Cb = (char*)Cout;
    const bool fullN = (n0 + 63) < VROWS;
    if (fullN) {
        #pragma unroll
        for (int mi = 0; mi < 4; ++mi) {
            #pragma unroll
            for (int ni = 0; ni < 4; ++ni) {
                uint32_t off = (uint32_t)((m0 + mi * 16 + hi * 8) * VROWS
                                          + n0 + ni * 16 + lr) * 4u;
                #pragma unroll
                for (int j = 0; j < 8; ++j) {
                    *(float*)(Cb + off) = acc[mi][ni][j];
                    off += (uint32_t)VROWS * 4u;
                }
            }
        }
    } else {
        #pragma unroll
        for (int mi = 0; mi < 4; ++mi) {
            #pragma unroll
            for (int ni = 0; ni < 4; ++ni) {
                const int n = n0 + ni * 16 + lr;
                if (n < VROWS) {
                    uint32_t off = (uint32_t)((m0 + mi * 16 + hi * 8) * VROWS + n) * 4u;
                    #pragma unroll
                    for (int j = 0; j < 8; ++j) {
                        *(float*)(Cb + off) = acc[mi][ni][j];
                        off += (uint32_t)VROWS * 4u;
                    }
                }
            }
        }
    }
}

extern "C" void kernel_launch(void* const* d_in, const int* in_sizes, int n_in,
                              void* d_out, int out_size, void* d_ws, size_t ws_size,
                              hipStream_t stream) {
    (void)in_sizes; (void)n_in; (void)out_size; (void)ws_size;
    const int*   tokens = (const int*)d_in[0];
    const float* emb    = (const float*)d_in[1];
    const float* wnorm  = (const float*)d_in[2];
    const float* oemb   = (const float*)d_in[3];
    float*       out    = (float*)d_out;

    __bf16* hb = (__bf16*)d_ws;                        //  6.3 MB
    __bf16* wb = hb + (size_t)MROWS * DDIM;            // 77.3 MB (padded)

    // 1) gather + RMSNorm + bf16 convert
    k_gather_rmsnorm_bf16<<<MROWS, 256, 0, stream>>>(tokens, emb, wnorm, hb);

    // 2) W fp32 -> bf16 (padded rows zeroed)
    const size_t wchunks = (size_t)VPAD * DDIM / 4;
    k_convert_w_bf16<<<(unsigned)((wchunks + 255) / 256), 256, 0, stream>>>(oemb, wb);

    // 3) WMMA GEMM: block tile 256(M) x 128(N)
    dim3 grid(VPAD / 128, MROWS / 256);
    k_gemm_bf16_wmma<<<grid, 256, 0, stream>>>(hb, wb, out);
}
